// LSTMLayer_4655744549337
// MI455X (gfx1250) — matrix operand order repaired and checked by reading, not compiled, not run on hardware
//
#include <hip/hip_runtime.h>
#include <hip/hip_bf16.h>
#include <stdint.h>

typedef __attribute__((ext_vector_type(16))) __bf16 v16bf;
typedef __attribute__((ext_vector_type(8)))  __bf16 v8bf;
typedef __attribute__((ext_vector_type(8)))  float  v8f;
typedef __attribute__((ext_vector_type(4)))  float  v4f;
typedef __attribute__((ext_vector_type(4)))  unsigned int v4u;

#define T_SEQ 512
#define BATCH 64
#define DDIM  1024
#define HDIM  1024
#define G4H   4096

// round-to-nearest-even fp32 -> bf16 (as raw u16)
static __device__ inline unsigned f2bf_u(float f) {
  unsigned u = __float_as_uint(f);
  unsigned r = 0x7FFFu + ((u >> 16) & 1u);
  return (u + r) >> 16;
}
static __device__ inline unsigned short f2bf(float f) { return (unsigned short)f2bf_u(f); }
static __device__ inline unsigned pack2bf(float lo, float hi) {
  return f2bf_u(lo) | (f2bf_u(hi) << 16);
}

// gfx1250 async global->LDS DMA (ASYNCcnt-tracked), 16B per lane
static __device__ inline void async_ld_b128(unsigned lds_off, const void* gptr) {
  asm volatile("global_load_async_to_lds_b128 %0, %1, off"
               :: "v"(lds_off), "v"((unsigned long long)(uintptr_t)gptr)
               : "memory");
}
static __device__ inline void wait_async0() {
  asm volatile("s_wait_asynccnt 0x0" ::: "memory");
}

// ---------------------------------------------------------------------------
// Prep: pack [D,H] fp32 gate matrices into transposed bf16 [4H][D] (gate-major
// N = gate*1024 + col). Gate order matches reference concat: i, f, o, c.
// ---------------------------------------------------------------------------
__global__ __launch_bounds__(256)
void pack_wt_kernel(const float* __restrict__ s0, const float* __restrict__ s1,
                    const float* __restrict__ s2, const float* __restrict__ s3,
                    unsigned short* __restrict__ dst) {
  size_t idx = (size_t)blockIdx.x * 256 + threadIdx.x;   // over 4096*1024
  int k   = (int)(idx & 1023);
  int n   = (int)(idx >> 10);
  int g   = n >> 10;
  int col = n & 1023;
  const float* s = (g == 0) ? s0 : (g == 1) ? s1 : (g == 2) ? s2 : s3;
  dst[idx] = f2bf(s[(size_t)k * 1024 + col]);
}

__global__ __launch_bounds__(256)
void prep_misc_kernel(const float* __restrict__ bi, const float* __restrict__ bff,
                      const float* __restrict__ bo, const float* __restrict__ bc,
                      const float* __restrict__ s0,
                      float* __restrict__ bias, unsigned short* __restrict__ hb,
                      unsigned* __restrict__ bar) {
  int idx = blockIdx.x * 256 + threadIdx.x;              // 65536 threads
  if (idx < 4096) {
    int g = idx >> 10, c = idx & 1023;
    const float* s = (g == 0) ? bi : (g == 1) ? bff : (g == 2) ? bo : bc;
    bias[idx] = s[c];
  }
  if (idx < BATCH * HDIM) hb[idx] = f2bf(s0[idx]);       // h0 into buffer 0
  if (idx < 4) bar[idx] = 0;
}

// ---------------------------------------------------------------------------
// Phase 1: xW[T*B, 4H] = x[T*B, D] @ W[D, 4H] + b   (bf16 WMMA, fp32 accum)
// Block: 256 thr = 8 waves, tile 128(M) x 64(N); wave = 64(M) x 16(N).
// ---------------------------------------------------------------------------
__global__ __launch_bounds__(256)
void lstm_xw_gemm(const float* __restrict__ x, const unsigned short* __restrict__ WbT,
                  const float* __restrict__ bias, float* __restrict__ xW) {
  __shared__ unsigned short As[128 * 40];                // row stride 40 halves (80B)
  const int tid  = threadIdx.x;
  const int lane = tid & 31;
  const int w    = tid >> 5;
  const int wr   = w >> 2;                               // 0..1  M half
  const int wn   = w & 3;                                // 0..3  N tile
  const int m0   = blockIdx.y * 128;
  const int n0   = blockIdx.x * 64;

  v8f acc[4];
  #pragma unroll
  for (int i = 0; i < 4; ++i)
    #pragma unroll
    for (int j = 0; j < 8; ++j) acc[i][j] = 0.0f;

  const int arow  = tid >> 1;                            // 0..127
  const int ahalf = tid & 1;                             // k chunk 0/16
  const int kb    = (lane >> 4) * 8;                     // A frag K base (halves)

  for (int k0 = 0; k0 < DDIM; k0 += 32) {
    __syncthreads();
    { // stage A tile 128x32: fp32 -> bf16 into LDS (packed 32-bit writes)
      const float* xs = x + (size_t)(m0 + arow) * DDIM + k0 + ahalf * 16;
      v4f f0 = *(const v4f*)(xs + 0);
      v4f f1 = *(const v4f*)(xs + 4);
      v4f f2 = *(const v4f*)(xs + 8);
      v4f f3 = *(const v4f*)(xs + 12);
      v4u p0, p1;
      p0[0] = pack2bf(f0[0], f0[1]); p0[1] = pack2bf(f0[2], f0[3]);
      p0[2] = pack2bf(f1[0], f1[1]); p0[3] = pack2bf(f1[2], f1[3]);
      p1[0] = pack2bf(f2[0], f2[1]); p1[1] = pack2bf(f2[2], f2[3]);
      p1[2] = pack2bf(f3[0], f3[1]); p1[3] = pack2bf(f3[2], f3[3]);
      unsigned short* d = As + arow * 40 + ahalf * 16;
      *(v4u*)(d) = p0; *(v4u*)(d + 8) = p1;
    }
    __syncthreads();

    // B fragment straight from global WbT[N][K]: halves kb..kb+7 and kb+16..kb+23 per lane
    const unsigned short* bp =
        WbT + (size_t)(n0 + wn * 16 + (lane & 15)) * DDIM + k0 + (lane >> 4) * 8;
    v16bf bfrag;
    {
      v8bf blo = *(const v8bf*)(bp);
      v8bf bhi = *(const v8bf*)(bp + 16);
      #pragma unroll
      for (int i = 0; i < 8; ++i) { bfrag[i] = blo[i]; bfrag[8 + i] = bhi[i]; }
    }
    #pragma unroll
    for (int mt = 0; mt < 4; ++mt) {
      const unsigned short* ap = As + (wr * 64 + mt * 16 + (lane & 15)) * 40;
      v8bf alo = *(const v8bf*)(ap + kb);
      v8bf ahi = *(const v8bf*)(ap + kb + 16);
      v16bf afrag;
      #pragma unroll
      for (int i = 0; i < 8; ++i) { afrag[i] = alo[i]; afrag[8 + i] = ahi[i]; }
      acc[mt] = __builtin_amdgcn_wmma_f32_16x16x32_bf16(
          false, afrag, false, bfrag, (short)0, acc[mt], false, false);
    }
  }

  const int col = n0 + wn * 16 + (lane & 15);
  const float bv = bias[col];
  #pragma unroll
  for (int mt = 0; mt < 4; ++mt)
    #pragma unroll
    for (int v = 0; v < 8; ++v) {
      int row = m0 + wr * 64 + mt * 16 + v + 8 * (lane >> 4);
      xW[(size_t)row * G4H + col] = acc[mt][v] + bv;
    }
}

// ---------------------------------------------------------------------------
// Grid-wide barrier (all 64 persistent blocks co-resident)
// ---------------------------------------------------------------------------
static __device__ inline void grid_barrier(unsigned* bar) {
  __syncthreads();
  if (threadIdx.x == 0) {
    __threadfence();
    volatile unsigned* gen = bar + 1;
    unsigned g = *gen;
    unsigned prev = atomicAdd(bar, 1u);
    if (prev == gridDim.x - 1) {
      bar[0] = 0;
      __threadfence();
      atomicAdd((unsigned*)(bar + 1), 1u);
    } else {
      while (*gen == g) { __builtin_amdgcn_s_sleep(2); }
    }
    __threadfence();
  }
  __syncthreads();
}

// ---------------------------------------------------------------------------
// Phase 2: persistent recurrent kernel. 64 blocks x 128 thr (4 waves).
// Block owns H-columns [cg*16, cg*16+16) for ALL 4 gates (wave = gate).
// h tile (A matrix) is DMA'd into LDS with async global->LDS copies, double
// buffered in K-chunks of 128 so the DMA of chunk j+1 overlaps the 16 WMMAs
// of chunk j. Cell state c lives in registers for the whole sequence.
// ---------------------------------------------------------------------------
#define KCH   128                      // K chunk (halves)
#define ASTR  136                      // LDS row stride in halves (272B, 16B-aligned)

__global__ __launch_bounds__(128)
void lstm_recurrent(const unsigned short* __restrict__ RbT,
                    const float* __restrict__ xW,
                    const float* __restrict__ c0,
                    unsigned short* __restrict__ hb,   // 2 x [B][H] bf16
                    float* __restrict__ out,           // [B][H] fp32
                    unsigned* __restrict__ bar) {
  __shared__ unsigned short Ah[2][64 * ASTR];          // 2 x 34KB? -> 2x17.4KB
  __shared__ float gbuf[4][64][16];                    // 16KB
  const int tid  = threadIdx.x;
  const int lane = tid & 31;
  const int gate = tid >> 5;                           // 0=i 1=f 2=o 3=c_hat
  const int cg   = blockIdx.x;                         // column group 0..63
  const size_t HB = (size_t)BATCH * HDIM;
  const int kb = (lane >> 4) * 8;

  const unsigned AhBase0 = (unsigned)(uintptr_t)(&Ah[0][0]);
  const unsigned AhBase1 = (unsigned)(uintptr_t)(&Ah[1][0]);

  // per-thread cell state: 8 elements of the 64x16 tile
  float c[8];
  #pragma unroll
  for (int j = 0; j < 8; ++j) {
    int e = tid * 8 + j;
    c[j] = c0[(size_t)(e >> 4) * HDIM + cg * 16 + (e & 15)];
  }

  const int gcol = gate * HDIM + cg * 16 + (lane & 15);
  const unsigned short* browBase = RbT + (size_t)gcol * HDIM; // row of R^T for this lane

  for (int t = 0; t < T_SEQ; ++t) {
    const unsigned short* hcur = hb + (size_t)(t & 1) * HB;
    unsigned short* hnxt = hb + (size_t)((t + 1) & 1) * HB;
    const float* xwrow = xW + (size_t)t * BATCH * G4H;

    // prefetch this step's xW tile rows into cache while we do the GEMM
    #pragma unroll
    for (int mt = 0; mt < 4; ++mt) {
      int m = mt * 16 + 8 * (lane >> 4);
      __builtin_prefetch(xwrow + (size_t)m * G4H + gcol, 0, 1);
    }

    v8f acc[4];
    #pragma unroll
    for (int i = 0; i < 4; ++i)
      #pragma unroll
      for (int j = 0; j < 8; ++j) acc[i][j] = 0.0f;

    // ---- issue async DMA of chunk 0 into buffer 0 ----
    #pragma unroll
    for (int i = 0; i < 8; ++i) {
      int cc = i * 128 + tid;                  // 1024 x 16B transfers per chunk
      int m = cc >> 4, q = cc & 15;
      async_ld_b128(AhBase0 + m * (ASTR * 2) + q * 16,
                    (const char*)hcur + (size_t)m * (HDIM * 2) + q * 16);
    }

    for (int ch = 0; ch < HDIM / KCH; ++ch) {  // 8 chunks
      wait_async0();                            // chunk ch landed (my wave's DMAs)
      __syncthreads();                          // all waves' DMAs visible

      // issue DMA of chunk ch+1 into the other buffer (overlaps WMMAs below)
      if (ch < HDIM / KCH - 1) {
        const unsigned nbuf = ((ch + 1) & 1) ? AhBase1 : AhBase0;
        const char* gsrc = (const char*)hcur + (size_t)(ch + 1) * KCH * 2;
        #pragma unroll
        for (int i = 0; i < 8; ++i) {
          int cc = i * 128 + tid;
          int m = cc >> 4, q = cc & 15;
          async_ld_b128(nbuf + m * (ASTR * 2) + q * 16,
                        gsrc + (size_t)m * (HDIM * 2) + q * 16);
        }
      }

      const unsigned short* Ab = &Ah[ch & 1][0];
      #pragma unroll
      for (int ks = 0; ks < KCH / 32; ++ks) {   // 4 x K=32 sub-steps
        const int kB = ch * KCH + ks * 32;
        // B fragment from RbT[N][K] (L2-resident): halves kb..kb+7 and kb+16..kb+23
        const unsigned short* bp = browBase + kB + (lane >> 4) * 8;
        v16bf bfrag;
        {
          v8bf blo = *(const v8bf*)(bp);
          v8bf bhi = *(const v8bf*)(bp + 16);
          #pragma unroll
          for (int i = 0; i < 8; ++i) { bfrag[i] = blo[i]; bfrag[8 + i] = bhi[i]; }
        }
        #pragma unroll
        for (int mt = 0; mt < 4; ++mt) {
          const unsigned short* ap = Ab + (mt * 16 + (lane & 15)) * ASTR + ks * 32;
          v8bf alo = *(const v8bf*)(ap + kb);
          v8bf ahi = *(const v8bf*)(ap + kb + 16);
          v16bf afrag;
          #pragma unroll
          for (int i = 0; i < 8; ++i) { afrag[i] = alo[i]; afrag[8 + i] = ahi[i]; }
          acc[mt] = __builtin_amdgcn_wmma_f32_16x16x32_bf16(
              false, afrag, false, bfrag, (short)0, acc[mt], false, false);
        }
      }
    }

    // g = acc + xW[t]; activate per gate; stage into LDS for cross-gate mix
    #pragma unroll
    for (int mt = 0; mt < 4; ++mt)
      #pragma unroll
      for (int v = 0; v < 8; ++v) {
        int m = mt * 16 + v + 8 * (lane >> 4);
        float g = acc[mt][v] + xwrow[(size_t)m * G4H + gcol];
        // reference: sigmoid(x) := (tanh(x)+1)/2 ; candidate uses tanh
        float a = (gate == 3) ? tanhf(g) : 0.5f * (tanhf(g) + 1.0f);
        gbuf[gate][m][lane & 15] = a;
      }
    __syncthreads();

    // elementwise cell/hidden update; write h_{t+1} bf16 to the other buffer
    #pragma unroll
    for (int j = 0; j < 8; ++j) {
      int e = tid * 8 + j;
      int m = e >> 4, n = e & 15;
      float iv = gbuf[0][m][n];
      float fv = gbuf[1][m][n];
      float ov = gbuf[2][m][n];
      float cv = gbuf[3][m][n];
      c[j] = fv * c[j] + iv * cv;
      float hv = ov * tanhf(c[j]);
      hnxt[(size_t)m * HDIM + cg * 16 + n] = f2bf(hv);
      if (t == T_SEQ - 1) out[(size_t)m * HDIM + cg * 16 + n] = hv;
    }
    __threadfence();
    grid_barrier(bar);
  }
}

// ---------------------------------------------------------------------------
// Launch
// d_in: 0:x 1:W_i 2:W_f 3:W_c 4:W_o 5:R_i 6:R_f 7:R_c 8:R_o
//       9:b_i 10:b_f 11:b_c 12:b_o 13:s0 14:c0
// ---------------------------------------------------------------------------
extern "C" void kernel_launch(void* const* d_in, const int* in_sizes, int n_in,
                              void* d_out, int out_size, void* d_ws, size_t ws_size,
                              hipStream_t stream) {
  const float* x  = (const float*)d_in[0];
  const float* Wi = (const float*)d_in[1];
  const float* Wf = (const float*)d_in[2];
  const float* Wc = (const float*)d_in[3];
  const float* Wo = (const float*)d_in[4];
  const float* Ri = (const float*)d_in[5];
  const float* Rf = (const float*)d_in[6];
  const float* Rc = (const float*)d_in[7];
  const float* Ro = (const float*)d_in[8];
  const float* bi = (const float*)d_in[9];
  const float* bf = (const float*)d_in[10];
  const float* bc = (const float*)d_in[11];
  const float* bo = (const float*)d_in[12];
  const float* s0 = (const float*)d_in[13];
  const float* c0 = (const float*)d_in[14];

  char* ws = (char*)d_ws;
  size_t off = 0;
  float* xW = (float*)(ws + off);            off += (size_t)T_SEQ * BATCH * G4H * 4;  // 512 MB
  unsigned short* WbT = (unsigned short*)(ws + off); off += (size_t)G4H * DDIM * 2;   // 8 MB
  unsigned short* RbT = (unsigned short*)(ws + off); off += (size_t)G4H * HDIM * 2;   // 8 MB
  float* bias = (float*)(ws + off);          off += G4H * 4;
  unsigned short* hb = (unsigned short*)(ws + off); off += (size_t)2 * BATCH * HDIM * 2;
  unsigned* bar = (unsigned*)(ws + off);     off += 256;

  // pack weights (gate order i,f,o,c to match reference concat)
  pack_wt_kernel<<<(G4H * DDIM) / 256, 256, 0, stream>>>(Wi, Wf, Wo, Wc, WbT);
  pack_wt_kernel<<<(G4H * HDIM) / 256, 256, 0, stream>>>(Ri, Rf, Ro, Rc, RbT);
  prep_misc_kernel<<<(BATCH * HDIM) / 256, 256, 0, stream>>>(bi, bf, bo, bc, s0, bias, hb, bar);

  // phase 1: big input-projection GEMM
  dim3 g1(G4H / 64, (T_SEQ * BATCH) / 128);
  lstm_xw_gemm<<<g1, 256, 0, stream>>>(x, WbT, bias, xW);

  // phase 2: persistent recurrence over 512 steps
  lstm_recurrent<<<HDIM / 16, 128, 0, stream>>>(RbT, xW, c0, hb, (float*)d_out, bar);
}
